// Attention_8469675507941
// MI455X (gfx1250) — matrix operand order repaired and checked by reading, not compiled, not run on hardware
//
#include <hip/hip_runtime.h>

typedef __attribute__((ext_vector_type(16))) _Float16 v16h;
typedef __attribute__((ext_vector_type(8)))  float    v8f;

#define DIM    768
#define NH     12
#define DH     64
#define SEQ    1024
#define NB     8
#define NROWS  (NB * SEQ)        // 8192
#define QKV_N  (3 * NH * DH)     // 2304

// ---- CDNA5 async global->LDS copy (guarded; falls back to sync copy) ----
#if defined(__gfx1250__) && __has_builtin(__builtin_amdgcn_global_load_async_to_lds_b128)
#define HAVE_ASYNC 1
#else
#define HAVE_ASYNC 0
#endif

#if HAVE_ASYNC
typedef int v4i_vs __attribute__((vector_size(4 * sizeof(int))));
typedef __attribute__((address_space(1))) v4i_vs* async_gptr;
typedef __attribute__((address_space(3))) v4i_vs* async_lptr;
#endif

// copy 8 f16 (16 bytes) from global to LDS
__device__ __forceinline__ void copy16_async(const _Float16* g, _Float16* l) {
#if HAVE_ASYNC
  __builtin_amdgcn_global_load_async_to_lds_b128((async_gptr)g, (async_lptr)l, 0, 0);
#else
  *(uint4*)l = *(const uint4*)g;
#endif
}

__device__ __forceinline__ void wait_async() {
#if HAVE_ASYNC
#if __has_builtin(__builtin_amdgcn_s_wait_asynccnt)
  __builtin_amdgcn_s_wait_asynccnt(0);
#else
  asm volatile("s_wait_asynccnt 0" ::: "memory");
#endif
#endif
}

// ---- WMMA fragment loaders (wave32, 16x16x32 f16) ----------------------
// A-matrix 16x32 (MxK): lane L -> M = L%16 (+tile), half = L/16.
//   halves 0..15 -> K = kbase + 16*half + i
__device__ __forceinline__ v16h ld_afrag(const _Float16* s, int ld, int m,
                                         int kbase, int half) {
  const _Float16* p0 = s + m * ld + kbase + 16 * half;
  v16h a;
#pragma unroll
  for (int i = 0; i < 16; ++i) a[i] = p0[i];
  return a;
}
// B-matrix 32x16 (KxN): lane L -> N = L%16 (+tile), half = L/16.
//   halves 0..15 -> K = kbase + 16*half + i ; LDS tile stored [n][k].
__device__ __forceinline__ v16h ld_bfrag(const _Float16* s, int ld, int n,
                                         int kbase, int half) {
  const _Float16* p = s + n * ld + kbase + 16 * half;
  v16h b;
#pragma unroll
  for (int i = 0; i < 16; ++i) b[i] = p[i];
  return b;
}

#define WMMA_F16(A, B, C) \
  __builtin_amdgcn_wmma_f32_16x16x32_f16(false, (A), false, (B), (short)0, (C), false, false)

// ------------------------------------------------------------------------
// Kernel 1: qkv = x @ W_qkv^T, scatter to Q/K/V f16 [B,h,T,64]
// grid (NROWS/128, QKV_N/128), block 256 (8 waves = 4x2).
// 128x128 tile, BK=32; wave sub-tile 32x64 -> 8 WMMA / wave / K-step.
// ------------------------------------------------------------------------
__global__ __launch_bounds__(256) void qkv_gemm(const float* __restrict__ x,
                                                const float* __restrict__ w,
                                                _Float16* __restrict__ Qf,
                                                _Float16* __restrict__ Kf,
                                                _Float16* __restrict__ Vf) {
  __shared__ alignas(16) _Float16 As[128 * 32];
  __shared__ alignas(16) _Float16 Bs[128 * 32];
  const int m0 = blockIdx.x * 128, n0 = blockIdx.y * 128;
  const int tid = threadIdx.x;
  const int wave = tid >> 5, lane = tid & 31, half = lane >> 4, ln = lane & 15;
  const int waveM = wave >> 1, waveN = wave & 1;
  const int lrow = tid >> 1, lk = (tid & 1) * 16;

  v8f acc[2][4] = {};
  for (int kk = 0; kk < DIM; kk += 32) {
    __syncthreads();
    const float4* ga = (const float4*)(x + (size_t)(m0 + lrow) * DIM + kk + lk);
    const float4* gb = (const float4*)(w + (size_t)(n0 + lrow) * DIM + kk + lk);
#pragma unroll
    for (int v = 0; v < 4; ++v) {
      float4 av = ga[v], bv = gb[v];
      _Float16* pa = As + lrow * 32 + lk + 4 * v;
      _Float16* pb = Bs + lrow * 32 + lk + 4 * v;
      pa[0] = (_Float16)av.x; pa[1] = (_Float16)av.y;
      pa[2] = (_Float16)av.z; pa[3] = (_Float16)av.w;
      pb[0] = (_Float16)bv.x; pb[1] = (_Float16)bv.y;
      pb[2] = (_Float16)bv.z; pb[3] = (_Float16)bv.w;
    }
    __syncthreads();
    v16h a0 = ld_afrag(As, 32, 32 * waveM + ln, 0, half);
    v16h a1 = ld_afrag(As, 32, 32 * waveM + 16 + ln, 0, half);
#pragma unroll
    for (int j = 0; j < 4; ++j) {
      v16h b = ld_bfrag(Bs, 32, 64 * waveN + 16 * j + ln, 0, half);
      acc[0][j] = WMMA_F16(a0, b, acc[0][j]);
      acc[1][j] = WMMA_F16(a1, b, acc[1][j]);
    }
  }
  // epilogue: scatter to Q/K/V [B,h,T,64] f16
#pragma unroll
  for (int sm = 0; sm < 2; ++sm) {
    const int mbase = m0 + 32 * waveM + 16 * sm + 8 * half;
#pragma unroll
    for (int j = 0; j < 4; ++j) {
      const int n = n0 + 64 * waveN + 16 * j + ln;
      const int sel = n / (NH * DH);
      const int rem = n % (NH * DH);
      const int head = rem / DH, dd = rem % DH;
      _Float16* dst = (sel == 0) ? Qf : ((sel == 1) ? Kf : Vf);
#pragma unroll
      for (int r = 0; r < 8; ++r) {
        const int m = mbase + r;
        const int bb = m / SEQ, t = m % SEQ;
        dst[((size_t)(bb * NH + head) * SEQ + t) * DH + dd] = (_Float16)acc[sm][j][r];
      }
    }
  }
}

// ------------------------------------------------------------------------
// Kernel 2: flash attention, grid (SEQ/64, NH, NB), block 128 (4 waves).
// Double-buffered: next K tile via async global->LDS, next V tile via
// register prefetch + transposed LDS store; overlap with WMMA compute.
// logits = (q.k)/SCALE, SCALE = 1/8 -> multiply by 8 (faithful).
// ------------------------------------------------------------------------
__global__ __launch_bounds__(128) void attn(const _Float16* __restrict__ Qf,
                                            const _Float16* __restrict__ Kf,
                                            const _Float16* __restrict__ Vf,
                                            _Float16* __restrict__ Ocat) {
  __shared__ alignas(16) _Float16 Qs[64 * 64];
  __shared__ alignas(16) _Float16 Ks[2][64 * 64];
  __shared__ alignas(16) _Float16 Vt[2][64 * 64];  // transposed: [d][s]
  __shared__ alignas(16) _Float16 Ps[64 * 64];
  const int q0 = blockIdx.x * 64;
  const int head = blockIdx.y, bat = blockIdx.z;
  const size_t base = (size_t)(bat * NH + head) * SEQ * DH;
  const _Float16* Qp = Qf + base;
  const _Float16* Kp = Kf + base;
  const _Float16* Vp = Vf + base;

  const int tid = threadIdx.x;
  const int wave = tid >> 5, lane = tid & 31, half = lane >> 4, ln = lane & 15;
  const int lrow = tid >> 1, lc = (tid & 1) * 32;

  // ---- preload: Q tile + K/V tile 0 ----
  {
    const _Float16* g = Qp + (size_t)(q0 + lrow) * DH + lc;
    _Float16* s = Qs + lrow * 64 + lc;
#pragma unroll
    for (int v = 0; v < 4; ++v) copy16_async(g + 8 * v, s + 8 * v);
  }
  {
    const _Float16* g = Kp + (size_t)lrow * DH + lc;
    _Float16* s = Ks[0] + lrow * 64 + lc;
#pragma unroll
    for (int v = 0; v < 4; ++v) copy16_async(g + 8 * v, s + 8 * v);
  }
  {
    const unsigned int* gv = (const unsigned int*)(Vp + (size_t)lrow * DH + lc);
    unsigned short* vt = (unsigned short*)Vt[0];
#pragma unroll
    for (int i = 0; i < 16; ++i) {
      const unsigned int u = gv[i];
      const int d = lc + 2 * i;
      vt[d * 64 + lrow] = (unsigned short)(u & 0xffffu);
      vt[(d + 1) * 64 + lrow] = (unsigned short)(u >> 16);
    }
  }
  wait_async();
  __syncthreads();

  const v16h qf0 = ld_afrag(Qs, 64, 16 * wave + ln, 0, half);
  const v16h qf1 = ld_afrag(Qs, 64, 16 * wave + ln, 32, half);

  v8f oacc[4] = {};
  float mrow[8], lsum[8];
#pragma unroll
  for (int r = 0; r < 8; ++r) { mrow[r] = -3.0e38f; lsum[r] = 0.0f; }

  for (int it = 0; it < SEQ / 64; ++it) {
    const int cur = it & 1, nxt = cur ^ 1;
    const _Float16* KsC = Ks[0] + cur * 64 * 64;
    const _Float16* VtC = Vt[0] + cur * 64 * 64;

    // ---- prefetch next tile (overlaps with compute below) ----
    unsigned int vr[16];
    const bool pre = (it + 1) < SEQ / 64;
    if (pre) {
      const int kb = (it + 1) * 64;
      const _Float16* g = Kp + (size_t)(kb + lrow) * DH + lc;
      _Float16* s = Ks[0] + nxt * 64 * 64 + lrow * 64 + lc;
#pragma unroll
      for (int v = 0; v < 4; ++v) copy16_async(g + 8 * v, s + 8 * v);
      const unsigned int* gv = (const unsigned int*)(Vp + (size_t)(kb + lrow) * DH + lc);
#pragma unroll
      for (int i = 0; i < 16; ++i) vr[i] = gv[i];
    }

    // ---- S = Q @ K^T (16 rows x 64 keys per wave) ----
    v8f sac[4];
#pragma unroll
    for (int j = 0; j < 4; ++j) {
      v16h b0 = ld_bfrag(KsC, 64, 16 * j + ln, 0, half);
      v16h b1 = ld_bfrag(KsC, 64, 16 * j + ln, 32, half);
      v8f z = {};
      z = WMMA_F16(qf0, b0, z);
      sac[j] = WMMA_F16(qf1, b1, z);
    }

    // ---- online softmax (row m = r + 8*half, N across 16-lane half) ----
#pragma unroll
    for (int r = 0; r < 8; ++r) {
      float rm = fmaxf(fmaxf(sac[0][r], sac[1][r]), fmaxf(sac[2][r], sac[3][r]));
      rm *= 8.0f;
#pragma unroll
      for (int msk = 1; msk < 16; msk <<= 1)
        rm = fmaxf(rm, __shfl_xor(rm, msk, 32));
      const float mnew = fmaxf(mrow[r], rm);
      const float corr = __expf(mrow[r] - mnew);
      float ps = 0.0f;
      const int prow = 16 * wave + r + 8 * half;
#pragma unroll
      for (int j = 0; j < 4; ++j) {
        const float p = __expf(sac[j][r] * 8.0f - mnew);
        ps += p;
        Ps[prow * 64 + 16 * j + ln] = (_Float16)p;
        oacc[j][r] *= corr;
      }
#pragma unroll
      for (int msk = 1; msk < 16; msk <<= 1) ps += __shfl_xor(ps, msk, 32);
      lsum[r] = lsum[r] * corr + ps;
      mrow[r] = mnew;
    }
    __syncthreads();  // Ps visible across lanes of the workgroup

    // ---- O += P @ V ----
    v16h pf0 = ld_afrag(Ps, 64, 16 * wave + ln, 0, half);
    v16h pf1 = ld_afrag(Ps, 64, 16 * wave + ln, 32, half);
#pragma unroll
    for (int j = 0; j < 4; ++j) {
      v16h vb0 = ld_bfrag(VtC, 64, 16 * j + ln, 0, half);
      v16h vb1 = ld_bfrag(VtC, 64, 16 * j + ln, 32, half);
      oacc[j] = WMMA_F16(pf0, vb0, oacc[j]);
      oacc[j] = WMMA_F16(pf1, vb1, oacc[j]);
    }

    // ---- write prefetched V regs into the other buffer (transposed) ----
    if (pre) {
      unsigned short* vt = (unsigned short*)(Vt[0] + nxt * 64 * 64);
#pragma unroll
      for (int i = 0; i < 16; ++i) {
        const int d = lc + 2 * i;
        vt[d * 64 + lrow] = (unsigned short)(vr[i] & 0xffffu);
        vt[(d + 1) * 64 + lrow] = (unsigned short)(vr[i] >> 16);
      }
    }
    wait_async();
    __syncthreads();  // next buffers ready; Ps free to overwrite
  }

  // ---- normalize + write O concat [B*T, 768] f16 ----
#pragma unroll
  for (int r = 0; r < 8; ++r) {
    const int trow = q0 + 16 * wave + r + 8 * half;
    const float inv = 1.0f / lsum[r];
#pragma unroll
    for (int j = 0; j < 4; ++j) {
      const int col = head * DH + 16 * j + ln;
      Ocat[((size_t)bat * SEQ + trow) * DIM + col] = (_Float16)(oacc[j][r] * inv);
    }
  }
}

// ------------------------------------------------------------------------
// Kernel 3: out = Ocat(f16) @ W_out^T, f32 out.
// grid (NROWS/128, DIM/128), block 256. A tile copied via async-to-LDS.
// ------------------------------------------------------------------------
__global__ __launch_bounds__(256) void out_gemm(const _Float16* __restrict__ A,
                                                const float* __restrict__ w,
                                                float* __restrict__ out) {
  __shared__ alignas(16) _Float16 As[128 * 32];
  __shared__ alignas(16) _Float16 Bs[128 * 32];
  const int m0 = blockIdx.x * 128, n0 = blockIdx.y * 128;
  const int tid = threadIdx.x;
  const int wave = tid >> 5, lane = tid & 31, half = lane >> 4, ln = lane & 15;
  const int waveM = wave >> 1, waveN = wave & 1;
  const int lrow = tid >> 1, lk = (tid & 1) * 16;

  v8f acc[2][4] = {};
  for (int kk = 0; kk < DIM; kk += 32) {
    __syncthreads();
    { // A tile already f16: async global->LDS copy (32 B / thread)
      const _Float16* g = A + (size_t)(m0 + lrow) * DIM + kk + lk;
      _Float16* s = As + lrow * 32 + lk;
      copy16_async(g, s);
      copy16_async(g + 8, s + 8);
    }
    { // W tile f32 -> f16
      const float4* gb = (const float4*)(w + (size_t)(n0 + lrow) * DIM + kk + lk);
#pragma unroll
      for (int v = 0; v < 4; ++v) {
        float4 bv = gb[v];
        _Float16* pb = Bs + lrow * 32 + lk + 4 * v;
        pb[0] = (_Float16)bv.x; pb[1] = (_Float16)bv.y;
        pb[2] = (_Float16)bv.z; pb[3] = (_Float16)bv.w;
      }
    }
    wait_async();
    __syncthreads();
    v16h a0 = ld_afrag(As, 32, 32 * waveM + ln, 0, half);
    v16h a1 = ld_afrag(As, 32, 32 * waveM + 16 + ln, 0, half);
#pragma unroll
    for (int j = 0; j < 4; ++j) {
      v16h b = ld_bfrag(Bs, 32, 64 * waveN + 16 * j + ln, 0, half);
      acc[0][j] = WMMA_F16(a0, b, acc[0][j]);
      acc[1][j] = WMMA_F16(a1, b, acc[1][j]);
    }
  }
#pragma unroll
  for (int sm = 0; sm < 2; ++sm) {
#pragma unroll
    for (int j = 0; j < 4; ++j) {
      const int n = n0 + 64 * waveN + 16 * j + ln;
#pragma unroll
      for (int r = 0; r < 8; ++r) {
        const int m = m0 + 32 * waveM + 16 * sm + r + 8 * half;
        out[(size_t)m * DIM + n] = acc[sm][j][r];
      }
    }
  }
}

// ------------------------------------------------------------------------
extern "C" void kernel_launch(void* const* d_in, const int* in_sizes, int n_in,
                              void* d_out, int out_size, void* d_ws, size_t ws_size,
                              hipStream_t stream) {
  (void)in_sizes; (void)n_in; (void)out_size; (void)ws_size;
  const float* x    = (const float*)d_in[0];
  const float* Wqkv = (const float*)d_in[1];
  const float* Wout = (const float*)d_in[2];
  float* out = (float*)d_out;

  _Float16* Qf = (_Float16*)d_ws;
  const size_t per = (size_t)NB * NH * SEQ * DH;  // 6,291,456 elements
  _Float16* Kf   = Qf + per;
  _Float16* Vf   = Kf + per;
  _Float16* Ocat = Vf + per;                      // NROWS*DIM elements

  qkv_gemm<<<dim3(NROWS / 128, QKV_N / 128), 256, 0, stream>>>(x, Wqkv, Qf, Kf, Vf);
  attn<<<dim3(SEQ / 64, NH, NB), 128, 0, stream>>>(Qf, Kf, Vf, Ocat);
  out_gemm<<<dim3(NROWS / 128, DIM / 128), 256, 0, stream>>>(Ocat, Wout, out);
}
